// GPT2Model_15857019257271
// MI455X (gfx1250) — matrix operand order repaired and checked by reading, not compiled, not run on hardware
//
#include <hip/hip_runtime.h>
#include <hip/hip_bf16.h>

typedef _Float16 f16;
typedef __attribute__((ext_vector_type(16))) _Float16 v16h;
typedef __attribute__((ext_vector_type(8)))  _Float16 v8h;
typedef __attribute__((ext_vector_type(4)))  _Float16 v4h;
typedef __attribute__((ext_vector_type(8)))  float    v8f;
typedef __attribute__((ext_vector_type(4)))  float    v4f;

#define TM 128
#define TN 128
#define TK 32

// ---------------------------------------------------------------------------
// Tiled WMMA GEMM with double-buffered LDS and b128 global staging:
//   C[M,N] = act(A[M,K] @ B + bias) (+ res)
// A, B fp32 in global; converted fp32->f16 while staging to LDS.
// transB=0: B is [K,N] row-major.  transB=1: B is [N,K] row-major (uses B^T).
// block = 256 threads (8 waves); tile 128x128x32; wave tile 32x64 (2x4 wmma).
// ---------------------------------------------------------------------------
__global__ __launch_bounds__(256) void gemm_wmma(
    const float* __restrict__ A, const float* __restrict__ B,
    const float* __restrict__ bias, const float* __restrict__ res,
    float* __restrict__ C, int M, int N, int K, int transB, int gelu)
{
    __shared__ __align__(16) f16 As[2][TM * TK];   // [m][k], stride 32 halves
    __shared__ __align__(16) f16 Bs[2][TN * TK];   // [n][k], stride 32 halves (B^T)

    const int tid  = threadIdx.x;
    const int lane = tid & 31;
    const int wave = tid >> 5;
    const int m0 = blockIdx.y * TM;
    const int n0 = blockIdx.x * TN;
    const int wm = (wave & 3) * 32;   // wave row base within tile
    const int wn = (wave >> 2) * 64;  // wave col base within tile
    const int lr = lane & 15;
    const int lh = lane >> 4;         // 0 or 1

    v8f acc[2][4] = {};
    v4f aR[4], bR[4];

    // ---- stage 1: issue b128 global loads for K-tile at k0 into registers
    auto issue = [&](int k0) {
        #pragma unroll
        for (int i = 0; i < 4; ++i) {                  // A: 128x32 = 1024 float4
            int q = tid + i * 256;
            int r = q >> 3, cq = (q & 7) * 4;
            aR[i] = *(const v4f*)&A[(size_t)(m0 + r) * K + (k0 + cq)];
        }
        #pragma unroll
        for (int i = 0; i < 4; ++i) {
            int q = tid + i * 256;
            if (!transB) {                             // B[K,N]: float4 along n
                int k = q >> 5, cq = (q & 31) * 4;
                int gn = n0 + cq;
                if (gn + 3 < N) {
                    bR[i] = *(const v4f*)&B[(size_t)(k0 + k) * N + gn];
                } else {
                    v4f t = {0.0f, 0.0f, 0.0f, 0.0f};
                    #pragma unroll
                    for (int j = 0; j < 4; ++j)
                        if (gn + j < N) t[j] = B[(size_t)(k0 + k) * N + gn + j];
                    bR[i] = t;
                }
            } else {                                   // B[N,K]: float4 along k
                int n = q >> 3, kq = (q & 7) * 4;
                int gn = n0 + n;
                v4f t = {0.0f, 0.0f, 0.0f, 0.0f};
                if (gn < N) t = *(const v4f*)&B[(size_t)gn * K + (k0 + kq)];
                bR[i] = t;
            }
        }
    };

    // ---- stage 2: convert to f16 and commit registers to LDS buffer `buf`
    auto commit = [&](int buf) {
        #pragma unroll
        for (int i = 0; i < 4; ++i) {
            int q = tid + i * 256;
            int r = q >> 3, cq = (q & 7) * 4;
            v4h h;
            #pragma unroll
            for (int j = 0; j < 4; ++j) h[j] = (f16)aR[i][j];
            *(v4h*)&As[buf][r * TK + cq] = h;
        }
        #pragma unroll
        for (int i = 0; i < 4; ++i) {
            int q = tid + i * 256;
            if (!transB) {                             // transpose into Bs[n][k]
                int k = q >> 5, cq = (q & 31) * 4;
                #pragma unroll
                for (int j = 0; j < 4; ++j)
                    Bs[buf][(cq + j) * TK + k] = (f16)bR[i][j];
            } else {                                   // already [n][k]
                int n = q >> 3, kq = (q & 7) * 4;
                v4h h;
                #pragma unroll
                for (int j = 0; j < 4; ++j) h[j] = (f16)bR[i][j];
                *(v4h*)&Bs[buf][n * TK + kq] = h;
            }
        }
    };

    const int KT = K / TK;
    issue(0);
    commit(0);
    __syncthreads();

    for (int kt = 0; kt < KT; ++kt) {
        const int cur = kt & 1;
        if (kt + 1 < KT) issue((kt + 1) * TK);         // loads in flight under WMMA

        // --- fragments + WMMA from LDS buffer `cur` ---
        v16h af[2];
        #pragma unroll
        for (int mi = 0; mi < 2; ++mi) {
            int m  = wm + mi * 16 + lr;
            int ko = lh * 8;
            v8h lo = *(const v8h*)&As[cur][m * TK + ko];
            v8h hi = *(const v8h*)&As[cur][m * TK + ko + 16];
            #pragma unroll
            for (int e = 0; e < 8; ++e) { af[mi][e] = lo[e]; af[mi][8 + e] = hi[e]; }
        }
        #pragma unroll
        for (int ni = 0; ni < 4; ++ni) {
            int n  = wn + ni * 16 + lr;
            int kb = lh * 8;
            v8h lo = *(const v8h*)&Bs[cur][n * TK + kb];
            v8h hi = *(const v8h*)&Bs[cur][n * TK + kb + 16];
            v16h bf;
            #pragma unroll
            for (int e = 0; e < 8; ++e) { bf[e] = lo[e]; bf[8 + e] = hi[e]; }
            #pragma unroll
            for (int mi = 0; mi < 2; ++mi) {
                acc[mi][ni] = __builtin_amdgcn_wmma_f32_16x16x32_f16(
                    false, af[mi], false, bf, (short)0, acc[mi][ni], false, false);
            }
        }

        if (kt + 1 < KT) commit(cur ^ 1);
        __syncthreads();
    }

    // --- epilogue: bias (+gelu) (+residual) ---
    #pragma unroll
    for (int mi = 0; mi < 2; ++mi) {
        #pragma unroll
        for (int ni = 0; ni < 4; ++ni) {
            int gcol = n0 + wn + ni * 16 + lr;
            if (gcol >= N) continue;
            float bv = bias ? bias[gcol] : 0.0f;
            #pragma unroll
            for (int r = 0; r < 8; ++r) {
                int grow = m0 + wm + mi * 16 + lh * 8 + r;
                float v = acc[mi][ni][r] + bv;
                if (gelu) v = 0.5f * v * (1.0f + erff(v * 0.70710678118654752f));
                if (res)  v += res[(size_t)grow * N + gcol];
                C[(size_t)grow * N + gcol] = v;
            }
        }
    }
}

// ---------------------------------------------------------------------------
// Causal attention, flash-style online softmax. One thread per query row.
// qkv: [B*T, 2304] fp32 (q|k|v). out: [B*T, 768] fp32.
// grid = (T/128, H=12, B=4), block = 128.
// ---------------------------------------------------------------------------
__global__ __launch_bounds__(128) void attn_kernel(
    const float* __restrict__ qkv, float* __restrict__ out)
{
    __shared__ __align__(16) float ks[32][64];
    __shared__ __align__(16) float vs[32][64];
    const int tid = threadIdx.x;
    const int qt  = blockIdx.x;
    const int h   = blockIdx.y;
    const int b   = blockIdx.z;
    const int t   = qt * 128 + tid;

    const size_t rowq = (size_t)(b * 1024 + t) * 2304 + h * 64;
    float q[64], y[64];
    #pragma unroll
    for (int i = 0; i < 16; ++i) {
        v4f tq = *(const v4f*)&qkv[rowq + i * 4];
        #pragma unroll
        for (int j = 0; j < 4; ++j) { q[i * 4 + j] = tq[j]; y[i * 4 + j] = 0.0f; }
    }

    float mi = -3.0e38f, li = 0.0f;
    const int ntile = qt * 4 + 4;   // keys 0 .. qt*128+127 in 32-key tiles

    for (int ktile = 0; ktile < ntile; ++ktile) {
        const int kst = ktile * 32;
        // cooperative K/V tile load, b128 accesses (32 keys x 64 dims each)
        #pragma unroll
        for (int i = 0; i < 4; ++i) {
            int idx = tid + i * 128;
            int kr = idx >> 4, d = (idx & 15) * 4;
            size_t rk = (size_t)(b * 1024 + kst + kr) * 2304 + h * 64 + d;
            *(v4f*)&ks[kr][d] = *(const v4f*)&qkv[rk + 768];
            *(v4f*)&vs[kr][d] = *(const v4f*)&qkv[rk + 1536];
        }
        __syncthreads();

        int rem  = t - kst + 1;
        int kmax = rem > 32 ? 32 : rem;   // causal mask; may be <= 0
        for (int kk = 0; kk < kmax; ++kk) {
            float s = 0.0f;
            #pragma unroll
            for (int d = 0; d < 64; ++d) s += q[d] * ks[kk][d];
            s *= 0.125f;                  // 1/sqrt(64)
            float mn   = fmaxf(mi, s);
            float corr = __expf(mi - mn);
            float p    = __expf(s - mn);
            li = li * corr + p;
            #pragma unroll
            for (int d = 0; d < 64; ++d) y[d] = y[d] * corr + p * vs[kk][d];
            mi = mn;
        }
        __syncthreads();
    }

    float inv = 1.0f / li;
    size_t ro = (size_t)(b * 1024 + t) * 768 + h * 64;
    #pragma unroll
    for (int i = 0; i < 16; ++i) {
        v4f tv;
        #pragma unroll
        for (int j = 0; j < 4; ++j) tv[j] = y[i * 4 + j] * inv;
        *(v4f*)&out[ro + i * 4] = tv;
    }
}

// ---------------------------------------------------------------------------
// LayerNorm over D=768. One block (256 thr) per row.
// ---------------------------------------------------------------------------
__global__ __launch_bounds__(256) void ln_kernel(
    const float* __restrict__ x, const float* __restrict__ g,
    const float* __restrict__ b, float* __restrict__ out)
{
    __shared__ float red[256];
    const int row = blockIdx.x;
    const float* xr = x + (size_t)row * 768;

    float s = 0.0f;
    #pragma unroll
    for (int j = 0; j < 3; ++j) s += xr[threadIdx.x + j * 256];
    red[threadIdx.x] = s; __syncthreads();
    for (int o = 128; o > 0; o >>= 1) {
        if (threadIdx.x < o) red[threadIdx.x] += red[threadIdx.x + o];
        __syncthreads();
    }
    float mu = red[0] * (1.0f / 768.0f);
    __syncthreads();

    float v = 0.0f;
    #pragma unroll
    for (int j = 0; j < 3; ++j) { float d = xr[threadIdx.x + j * 256] - mu; v += d * d; }
    red[threadIdx.x] = v; __syncthreads();
    for (int o = 128; o > 0; o >>= 1) {
        if (threadIdx.x < o) red[threadIdx.x] += red[threadIdx.x + o];
        __syncthreads();
    }
    float rstd = rsqrtf(red[0] * (1.0f / 768.0f) + 1e-5f);

    #pragma unroll
    for (int j = 0; j < 3; ++j) {
        int d = threadIdx.x + j * 256;
        out[(size_t)row * 768 + d] = (xr[d] - mu) * rstd * g[d] + b[d];
    }
}

// ---------------------------------------------------------------------------
// Token + position embedding. One block per row.
// ---------------------------------------------------------------------------
__global__ __launch_bounds__(256) void embed_kernel(
    const int* __restrict__ tokens, const float* __restrict__ wte,
    const float* __restrict__ wpe, float* __restrict__ x)
{
    const int row = blockIdx.x;       // b*1024 + t
    const int t   = row & 1023;
    const int tok = tokens[row];
    #pragma unroll
    for (int j = 0; j < 3; ++j) {
        int d = threadIdx.x + j * 256;
        x[(size_t)row * 768 + d] = wte[(size_t)tok * 768 + d] + wpe[(size_t)t * 768 + d];
    }
}

// ---------------------------------------------------------------------------
extern "C" void kernel_launch(void* const* d_in, const int* in_sizes, int n_in,
                              void* d_out, int out_size, void* d_ws, size_t ws_size,
                              hipStream_t stream) {
    const int*   tokens = (const int*)  d_in[0];
    const float* wte    = (const float*)d_in[1];
    const float* wpe    = (const float*)d_in[2];
    const float* ln1_g  = (const float*)d_in[3];
    const float* ln1_b  = (const float*)d_in[4];
    const float* wqkv   = (const float*)d_in[5];
    const float* bqkv   = (const float*)d_in[6];
    const float* wproj  = (const float*)d_in[7];
    const float* bproj  = (const float*)d_in[8];
    const float* ln2_g  = (const float*)d_in[9];
    const float* ln2_b  = (const float*)d_in[10];
    const float* wfc    = (const float*)d_in[11];
    const float* bfc    = (const float*)d_in[12];
    const float* wfc2   = (const float*)d_in[13];
    const float* bfc2   = (const float*)d_in[14];
    const float* lnf_g  = (const float*)d_in[15];
    const float* lnf_b  = (const float*)d_in[16];
    const float* w_head = (const float*)d_in[17];
    float* out = (float*)d_out;

    // workspace layout (fp32 elements)
    float* x   = (float*)d_ws;                       // 4096 x 768
    float* h   = x   + (size_t)4096 * 768;           // 4096 x 768
    float* qkv = h   + (size_t)4096 * 768;           // 4096 x 2304
    float* att = qkv + (size_t)4096 * 2304;          // 4096 x 768
    float* fc  = att + (size_t)4096 * 768;           // 4096 x 3072

    embed_kernel<<<4096, 256, 0, stream>>>(tokens, wte, wpe, x);

    for (int l = 0; l < 12; ++l) {
        ln_kernel<<<4096, 256, 0, stream>>>(x, ln1_g + l * 768, ln1_b + l * 768, h);
        gemm_wmma<<<dim3(18, 32), 256, 0, stream>>>(
            h, wqkv + (size_t)l * 768 * 2304, bqkv + l * 2304, nullptr, qkv,
            4096, 2304, 768, 0, 0);
        attn_kernel<<<dim3(8, 12, 4), 128, 0, stream>>>(qkv, att);
        gemm_wmma<<<dim3(6, 32), 256, 0, stream>>>(
            att, wproj + (size_t)l * 768 * 768, bproj + l * 768, x, x,
            4096, 768, 768, 0, 0);
        ln_kernel<<<4096, 256, 0, stream>>>(x, ln2_g + l * 768, ln2_b + l * 768, h);
        gemm_wmma<<<dim3(24, 32), 256, 0, stream>>>(
            h, wfc + (size_t)l * 768 * 3072, bfc + l * 3072, nullptr, fc,
            4096, 3072, 768, 0, 1);
        gemm_wmma<<<dim3(6, 32), 256, 0, stream>>>(
            fc, wfc2 + (size_t)l * 3072 * 768, bfc2 + l * 768, x, x,
            4096, 768, 3072, 0, 0);
    }

    ln_kernel<<<4096, 256, 0, stream>>>(x, lnf_g, lnf_b, h);
    // logits = h @ w_head^T   (w_head is [VOCAB, D] -> transB)
    gemm_wmma<<<dim3((50257 + 127) / 128, 32), 256, 0, stream>>>(
        h, w_head, nullptr, nullptr, out, 4096, 50257, 768, 1, 0);
}
